// AlexNetOWT_BN_69020124447416
// MI455X (gfx1250) — 8-bit matrix operand order repaired and checked by reading, compiled, not run on hardware
//
#include <hip/hip_runtime.h>
#include <stdint.h>

typedef __attribute__((ext_vector_type(8))) int v8i;
typedef __attribute__((ext_vector_type(4))) int v4i;

#define BN_EPS 1e-5f
#define NPART 240

__device__ __forceinline__ int a_kmap(int v, int grp, int b) {
  int base = (v & 3) * 4 + (v >> 2) * 32;
  return base + grp * 16 + b;
}
__device__ __forceinline__ int sgn_i(float x) { return (x > 0.f) ? 1 : ((x < 0.f) ? -1 : 0); }
__device__ __forceinline__ int imax2(int a, int b) { return a > b ? a : b; }

__global__ void k_sign_f32(const float* __restrict__ w, float* __restrict__ o, int n) {
  int i = blockIdx.x * 256 + threadIdx.x;
  if (i < n) o[i] = (float)sgn_i(w[i]);
}

__global__ void k_pack_conv_a(const float* __restrict__ w, int* __restrict__ wp, int ncot) {
  int id = blockIdx.x * 256 + threadIdx.x;
  int v = id & 7, lane = (id >> 3) & 31, rest = id >> 8;
  int cot = rest % ncot, k = rest / ncot;
  if (k >= 41) return;
  int grp = lane >> 4, co = cot * 16 + (lane & 15);
  unsigned word = 0;
  for (int b = 0; b < 4; ++b) {
    int ci = a_kmap(v, grp, b);
    int s = sgn_i(w[((size_t)co * 64 + ci) * 41 + k]);
    word |= ((unsigned)(s & 0xff)) << (8 * b);
  }
  wp[id] = (int)word;
}

__global__ void k_pack_fc_a(const float* __restrict__ w, int* __restrict__ wp,
                            int O, int F, int nOt) {
  int id = blockIdx.x * 256 + threadIdx.x;
  int v = id & 7, lane = (id >> 3) & 31, rest = id >> 8;
  int ot = rest % nOt, kt = rest / nOt;
  int grp = lane >> 4, o = ot * 16 + (lane & 15);
  unsigned word = 0;
  for (int b = 0; b < 4; ++b) {
    int f = kt * 64 + a_kmap(v, grp, b);
    int s = (o < O) ? sgn_i(w[(size_t)o * F + f]) : 0;
    word |= ((unsigned)(s & 0xff)) << (8 * b);
  }
  wp[id] = (int)word;
}

__global__ void k_bn0_stats(const float* __restrict__ x, const float* __restrict__ g,
                            const float* __restrict__ b, float* __restrict__ AB) {
  int c = blockIdx.x, tid = threadIdx.x;
  __shared__ float ss[256], qq[256];
  float s = 0.f, q = 0.f;
  for (int n = 0; n < 64; ++n) {
    const float* xp = x + (size_t)n * 48000 + (size_t)c * 16000;
    for (int l = tid; l < 16000; l += 256) { float v = xp[l]; s += v; q += v * v; }
  }
  ss[tid] = s; qq[tid] = q; __syncthreads();
  for (int st = 128; st > 0; st >>= 1) {
    if (tid < st) { ss[tid] += ss[tid + st]; qq[tid] += qq[tid + st]; }
    __syncthreads();
  }
  if (tid == 0) {
    float inv = 1.f / (64.f * 16000.f);
    float m = ss[0] * inv, var = qq[0] * inv - m * m;
    float a = g[c] * rsqrtf(var + BN_EPS);
    AB[c] = a; AB[3 + c] = b[c] - m * a;
  }
}

__global__ void k_bn_part64(const float* __restrict__ p, float* __restrict__ part, long R) {
  int b = blockIdx.x, tid = threadIdx.x;
  long chunk = (R + NPART - 1) / NPART;
  long r0 = (long)b * chunk; if (r0 > R) r0 = R;
  long r1 = r0 + chunk;      if (r1 > R) r1 = R;
  float s = 0.f, q = 0.f;
  const float* base = p + r0 * 64;
  long total = (r1 - r0) * 64;
  for (long j = tid; j < total; j += 256) { float v = base[j]; s += v; q += v * v; }
  __shared__ float ss[256], qq[256];
  ss[tid] = s; qq[tid] = q; __syncthreads();
  if (tid < 64) {
    float S = ss[tid] + ss[tid + 64] + ss[tid + 128] + ss[tid + 192];
    float Q = qq[tid] + qq[tid + 64] + qq[tid + 128] + qq[tid + 192];
    part[(size_t)b * 128 + tid] = S;
    part[(size_t)b * 128 + 64 + tid] = Q;
  }
}

__global__ void k_bn_final64(const float* __restrict__ part, const float* __restrict__ g,
                             const float* __restrict__ b, float* __restrict__ AB, long R) {
  int c = threadIdx.x;
  float S = 0.f, Q = 0.f;
  for (int i = 0; i < NPART; ++i) {
    S += part[(size_t)i * 128 + c];
    Q += part[(size_t)i * 128 + 64 + c];
  }
  float inv = 1.f / (float)R;
  float m = S * inv, var = Q * inv - m * m;
  float a = g[c] * rsqrtf(var + BN_EPS);
  AB[c] = a; AB[64 + c] = b[c] - m * a;
}

__global__ void k_bn_stats_cl(const float* __restrict__ p, const float* __restrict__ g,
                              const float* __restrict__ b, float* __restrict__ AB,
                              int C, long NL) {
  int c = blockIdx.x, tid = threadIdx.x;
  __shared__ float ss[256], qq[256];
  float s = 0.f, q = 0.f;
  for (long i = tid; i < NL; i += 256) { float v = p[i * (long)C + c]; s += v; q += v * v; }
  ss[tid] = s; qq[tid] = q; __syncthreads();
  for (int st = 128; st > 0; st >>= 1) {
    if (tid < st) { ss[tid] += ss[tid + st]; qq[tid] += qq[tid + st]; }
    __syncthreads();
  }
  if (tid == 0) {
    float inv = 1.f / (float)NL;
    float m = ss[0] * inv, var = qq[0] * inv - m * m;
    float a = g[c] * rsqrtf(var + BN_EPS);
    AB[c] = a; AB[C + c] = b[c] - m * a;
  }
}

__global__ void k_binarize_cl(const float* __restrict__ p, const float* __restrict__ AB,
                              int8_t* __restrict__ bin, int C, long total) {
  long i = (long)blockIdx.x * 256 + threadIdx.x;
  if (i < total) {
    int c = (int)(i % C);
    float v = fmaf(p[i], AB[c], AB[C + c]);
    bin[i] = (int8_t)sgn_i(v);
  }
}

__global__ void __launch_bounds__(256) k_conv1_pool(
    const float* __restrict__ x, const float* __restrict__ AB0,
    const float* __restrict__ w1s, const float* __restrict__ bias,
    float* __restrict__ pooled) {
  const int Lp = 5306;
  int id = blockIdx.x * 256 + threadIdx.x;
  int lp = id % Lp; int t = id / Lp; int co = t & 63; int n = t >> 6;
  if (n >= 64) return;
  const float Ac[3] = {AB0[0], AB0[1], AB0[2]};
  const float Bc[3] = {AB0[3], AB0[4], AB0[5]};
  const float* xn = x + (size_t)n * 48000;
  const float* wr = w1s + co * 123;
  float best = -3.0e38f;
  for (int d = 0; d < 3; ++d) {
    int l = lp * 3 + d;
    float s = 0.f;
    for (int ci = 0; ci < 3; ++ci) {
      const float* xc = xn + ci * 16000 + l;
      const float* wc = wr + ci * 41;
      float a = Ac[ci], bb = Bc[ci];
      for (int k = 0; k < 41; ++k) {
        float h = fminf(fmaxf(fmaf(xc[2 * k], a, bb), -1.f), 1.f);
        s = fmaf(h, wc[k], s);
      }
    }
    best = fmaxf(best, s);
  }
  pooled[(size_t)(n * Lp + lp) * 64 + co] = best + bias[co];
}

template <int NCOT>
__global__ void __launch_bounds__(64) k_conv_pool_iu8(
    const int8_t* __restrict__ bin, const int* __restrict__ wp,
    const float* __restrict__ bias, float* __restrict__ pooled,
    int Lin, int Lp) {
  const int Cout = NCOT * 16;
  int lane = threadIdx.x & 31;
  int wv = threadIdx.x >> 5;
  int n = blockIdx.y, t = blockIdx.x;
  int grp = lane >> 4, col = lane & 15;

  __shared__ int stage[4][48][16];

  v8i z = {0, 0, 0, 0, 0, 0, 0, 0};
  v8i acc[2][3];
  #pragma unroll
  for (int c = 0; c < 2; ++c)
    #pragma unroll
    for (int j = 0; j < 3; ++j) acc[c][j] = z;

  const int8_t* bbase = bin + (size_t)n * Lin * 64 + (size_t)(t * 48 + col) * 64 + grp * 16;
  const int* abase = wp + ((2 * wv) * 32 + lane) * 8;

  #pragma unroll
  for (int k = 0; k < 41; ++k) {
    const v4i* ap0 = (const v4i*)(abase + k * NCOT * 256);
    const v4i* ap1 = (const v4i*)(abase + k * NCOT * 256 + 256);
    v4i a0l = ap0[0], a0h = ap0[1];
    v4i a1l = ap1[0], a1h = ap1[1];
    v8i A0 = {a0l[0], a0l[1], a0l[2], a0l[3], a0h[0], a0h[1], a0h[2], a0h[3]};
    v8i A1 = {a1l[0], a1l[1], a1l[2], a1l[3], a1h[0], a1h[1], a1h[2], a1h[3]};
    const int8_t* bk = bbase + k * 128;
    #pragma unroll
    for (int j = 0; j < 3; ++j) {
      const v4i* bp = (const v4i*)(bk + j * 1024);
      v4i bl = bp[0], bh = bp[2];
      v8i B = {bl[0], bl[1], bl[2], bl[3], bh[0], bh[1], bh[2], bh[3]};
      acc[0][j] = __builtin_amdgcn_wmma_i32_16x16x64_iu8(true, A0, true, B, acc[0][j], false, false);
      acc[1][j] = __builtin_amdgcn_wmma_i32_16x16x64_iu8(true, A1, true, B, acc[1][j], false, false);
    }
  }

  #pragma unroll
  for (int c = 0; c < 2; ++c)
    #pragma unroll
    for (int j = 0; j < 3; ++j)
      #pragma unroll
      for (int r = 0; r < 8; ++r)
        stage[wv * 2 + c][j * 16 + col][r + 8 * grp] = acc[c][j][r];
  __syncthreads();

  #pragma unroll
  for (int i = 0; i < 16; ++i) {
    int e = lane + 32 * i;
    int ct = e >> 8;
    int rem = e & 255;
    int co_l = rem & 15, lpl = rem >> 4;
    int lp = t * 16 + lpl;
    if (lp < Lp) {
      int tile = wv * 2 + ct;
      int q0 = stage[tile][lpl * 3 + 0][co_l];
      int q1 = stage[tile][lpl * 3 + 1][co_l];
      int q2 = stage[tile][lpl * 3 + 2][co_l];
      int mx = imax2(q0, imax2(q1, q2));
      int co = tile * 16 + co_l;
      pooled[(size_t)(n * Lp + lp) * Cout + co] = (float)mx + bias[co];
    }
  }
}

template <int NKT, int NOT>
__global__ void __launch_bounds__(128) k_fc_iu8(
    const int8_t* __restrict__ bin, const int* __restrict__ wp,
    const float* __restrict__ bias, float* __restrict__ out,
    int F, int Ovalid) {
  int lane = threadIdx.x & 31;
  int wv = threadIdx.x >> 5;
  int ot = blockIdx.x;
  int grp = lane >> 4, col = lane & 15;
  int nIdx = wv * 16 + col;
  v8i z = {0, 0, 0, 0, 0, 0, 0, 0};
  v8i acc = z;
  const int* abase = wp + (ot * 32 + lane) * 8;
  const int8_t* bbase = bin + (size_t)nIdx * F + grp * 16;
  #pragma unroll
  for (int kt = 0; kt < NKT; ++kt) {
    const v4i* ap = (const v4i*)(abase + kt * NOT * 256);
    v4i al = ap[0], ah = ap[1];
    v8i A = {al[0], al[1], al[2], al[3], ah[0], ah[1], ah[2], ah[3]};
    const v4i* bp = (const v4i*)(bbase + kt * 64);
    v4i bl = bp[0], bh = bp[2];
    v8i B = {bl[0], bl[1], bl[2], bl[3], bh[0], bh[1], bh[2], bh[3]};
    acc = __builtin_amdgcn_wmma_i32_16x16x64_iu8(true, A, true, B, acc, false, false);
  }
  #pragma unroll
  for (int r = 0; r < 8; ++r) {
    int o = ot * 16 + r + 8 * grp;
    if (o < Ovalid) out[(size_t)nIdx * Ovalid + o] = (float)acc[r] + bias[o];
  }
}

__global__ void k_bin_fc1(const float* __restrict__ p5, int8_t* __restrict__ bin) {
  int i = blockIdx.x * 256 + threadIdx.x;
  if (i < 64 * 832) {
    int n = i / 832, f = i % 832;
    int c = f / 26, l = f % 26;
    float v = p5[(size_t)(n * 26 + l) * 32 + c];
    bin[i] = (int8_t)sgn_i(v);
  }
}

__global__ void k_logsoftmax(const float* __restrict__ y, const float* __restrict__ AB,
                             float* __restrict__ out) {
  int n = blockIdx.x, tid = threadIdx.x;
  __shared__ float red[256];
  __shared__ float sM, sLSE;
  const float* yn = y + (size_t)n * 1000;
  float lmax = -3.0e38f;
  for (int o = tid; o < 1000; o += 256)
    lmax = fmaxf(lmax, fmaf(yn[o], AB[o], AB[1000 + o]));
  red[tid] = lmax; __syncthreads();
  for (int st = 128; st > 0; st >>= 1) {
    if (tid < st) red[tid] = fmaxf(red[tid], red[tid + st]);
    __syncthreads();
  }
  if (tid == 0) sM = red[0];
  __syncthreads();
  float s = 0.f;
  for (int o = tid; o < 1000; o += 256)
    s += expf(fmaf(yn[o], AB[o], AB[1000 + o]) - sM);
  red[tid] = s; __syncthreads();
  for (int st = 128; st > 0; st >>= 1) {
    if (tid < st) red[tid] += red[tid + st];
    __syncthreads();
  }
  if (tid == 0) sLSE = sM + logf(red[0]);
  __syncthreads();
  for (int o = tid; o < 1000; o += 256)
    out[(size_t)n * 1000 + o] = fmaf(yn[o], AB[o], AB[1000 + o]) - sLSE;
}

extern "C" void kernel_launch(void* const* d_in, const int* in_sizes, int n_in,
                              void* d_out, int out_size, void* d_ws, size_t ws_size,
                              hipStream_t stream) {
  (void)in_sizes; (void)n_in; (void)out_size; (void)ws_size;
  const float* x     = (const float*)d_in[0];
  const float* bn0_g = (const float*)d_in[1];
  const float* bn0_b = (const float*)d_in[2];
  const float* w1    = (const float*)d_in[3];
  const float* c1b   = (const float*)d_in[4];
  const float* bn1_g = (const float*)d_in[5];
  const float* bn1_b = (const float*)d_in[6];
  const float* w2    = (const float*)d_in[7];
  const float* c2b   = (const float*)d_in[8];
  const float* bn2_g = (const float*)d_in[9];
  const float* bn2_b = (const float*)d_in[10];
  const float* w3    = (const float*)d_in[11];
  const float* c3b   = (const float*)d_in[12];
  const float* bn3_g = (const float*)d_in[13];
  const float* bn3_b = (const float*)d_in[14];
  const float* w4    = (const float*)d_in[15];
  const float* c4b   = (const float*)d_in[16];
  const float* bn4_g = (const float*)d_in[17];
  const float* bn4_b = (const float*)d_in[18];
  const float* w5    = (const float*)d_in[19];
  const float* c5b   = (const float*)d_in[20];
  const float* fc1w  = (const float*)d_in[21];
  const float* fc1b  = (const float*)d_in[22];
  const float* bn5_g = (const float*)d_in[23];
  const float* bn5_b = (const float*)d_in[24];
  const float* fc2w  = (const float*)d_in[25];
  const float* fc2b  = (const float*)d_in[26];
  const float* bn6_g = (const float*)d_in[27];
  const float* bn6_b = (const float*)d_in[28];
  float* out = (float*)d_out;

  char* ws = (char*)d_ws;
  size_t cur = 0;
  auto alloc = [&](size_t bytes) -> char* {
    cur = (cur + 255) & ~(size_t)255;
    char* p = ws + cur; cur += bytes; return p;
  };
  float*  w1s  = (float*)alloc(7872 * 4);
  int*    wp2  = (int*)alloc((size_t)41 * 4 * 256 * 4);
  int*    wp3  = (int*)alloc((size_t)41 * 4 * 256 * 4);
  int*    wp4  = (int*)alloc((size_t)41 * 4 * 256 * 4);
  int*    wp5  = (int*)alloc((size_t)41 * 2 * 256 * 4);
  int*    wpf1 = (int*)alloc((size_t)13 * 64 * 256 * 4);
  int*    wpf2 = (int*)alloc((size_t)16 * 63 * 256 * 4);
  float*  AB0  = (float*)alloc(6 * 4);
  float*  AB1  = (float*)alloc(128 * 4);
  float*  AB2  = (float*)alloc(128 * 4);
  float*  AB3  = (float*)alloc(128 * 4);
  float*  AB4  = (float*)alloc(128 * 4);
  float*  AB5  = (float*)alloc(2048 * 4);
  float*  AB6  = (float*)alloc(2000 * 4);
  float*  part = (float*)alloc((size_t)NPART * 128 * 4);
  float*  fc1o = (float*)alloc((size_t)64 * 1024 * 4);
  float*  fc2o = (float*)alloc((size_t)64 * 1000 * 4);
  int8_t* bF1  = (int8_t*)alloc(64 * 832);
  int8_t* bF2  = (int8_t*)alloc(64 * 1024);
  float*  poolA = (float*)alloc((size_t)64 * 5306 * 64 * 4);
  int8_t* binB  = (int8_t*)alloc((size_t)64 * 5306 * 64 + 128 * 64);

  k_sign_f32<<<(7872 + 255) / 256, 256, 0, stream>>>(w1, w1s, 7872);
  k_pack_conv_a<<<41 * 4, 256, 0, stream>>>(w2, wp2, 4);
  k_pack_conv_a<<<41 * 4, 256, 0, stream>>>(w3, wp3, 4);
  k_pack_conv_a<<<41 * 4, 256, 0, stream>>>(w4, wp4, 4);
  k_pack_conv_a<<<41 * 2, 256, 0, stream>>>(w5, wp5, 2);
  k_pack_fc_a<<<13 * 64, 256, 0, stream>>>(fc1w, wpf1, 1024, 832, 64);
  k_pack_fc_a<<<16 * 63, 256, 0, stream>>>(fc2w, wpf2, 1000, 1024, 63);

  k_bn0_stats<<<3, 256, 0, stream>>>(x, bn0_g, bn0_b, AB0);
  k_conv1_pool<<<(64 * 64 * 5306) / 256, 256, 0, stream>>>(x, AB0, w1s, c1b, poolA);

  long NL = (long)64 * 5306;
  k_bn_part64<<<NPART, 256, 0, stream>>>(poolA, part, NL);
  k_bn_final64<<<1, 64, 0, stream>>>(part, bn1_g, bn1_b, AB1, NL);
  k_binarize_cl<<<(int)((NL * 64 + 255) / 256), 256, 0, stream>>>(poolA, AB1, binB, 64, NL * 64);
  k_conv_pool_iu8<4><<<dim3(109, 64), 64, 0, stream>>>(binB, wp2, c2b, poolA, 5306, 1742);

  NL = (long)64 * 1742;
  k_bn_part64<<<NPART, 256, 0, stream>>>(poolA, part, NL);
  k_bn_final64<<<1, 64, 0, stream>>>(part, bn2_g, bn2_b, AB2, NL);
  k_binarize_cl<<<(int)((NL * 64 + 255) / 256), 256, 0, stream>>>(poolA, AB2, binB, 64, NL * 64);
  k_conv_pool_iu8<4><<<dim3(35, 64), 64, 0, stream>>>(binB, wp3, c3b, poolA, 1742, 554);

  NL = (long)64 * 554;
  k_bn_part64<<<NPART, 256, 0, stream>>>(poolA, part, NL);
  k_bn_final64<<<1, 64, 0, stream>>>(part, bn3_g, bn3_b, AB3, NL);
  k_binarize_cl<<<(int)((NL * 64 + 255) / 256), 256, 0, stream>>>(poolA, AB3, binB, 64, NL * 64);
  k_conv_pool_iu8<4><<<dim3(10, 64), 64, 0, stream>>>(binB, wp4, c4b, poolA, 554, 158);

  NL = (long)64 * 158;
  k_bn_part64<<<NPART, 256, 0, stream>>>(poolA, part, NL);
  k_bn_final64<<<1, 64, 0, stream>>>(part, bn4_g, bn4_b, AB4, NL);
  k_binarize_cl<<<(int)((NL * 64 + 255) / 256), 256, 0, stream>>>(poolA, AB4, binB, 64, NL * 64);
  k_conv_pool_iu8<2><<<dim3(2, 64), 32, 0, stream>>>(binB, wp5, c5b, poolA, 158, 26);

  k_bin_fc1<<<(64 * 832 + 255) / 256, 256, 0, stream>>>(poolA, bF1);
  k_fc_iu8<13, 64><<<64, 128, 0, stream>>>(bF1, wpf1, fc1b, fc1o, 832, 1024);
  k_bn_stats_cl<<<1024, 256, 0, stream>>>(fc1o, bn5_g, bn5_b, AB5, 1024, 64);
  k_binarize_cl<<<(64 * 1024 + 255) / 256, 256, 0, stream>>>(fc1o, AB5, bF2, 1024, (long)64 * 1024);

  k_fc_iu8<16, 63><<<63, 128, 0, stream>>>(bF2, wpf2, fc2b, fc2o, 1024, 1000);
  k_bn_stats_cl<<<1000, 256, 0, stream>>>(fc2o, bn6_g, bn6_b, AB6, 1000, 64);
  k_logsoftmax<<<64, 256, 0, stream>>>(fc2o, AB6, out);
}
